// VectorQuantizerEMA_24902220382359
// MI455X (gfx1250) — matrix operand order repaired and checked by reading, not compiled, not run on hardware
//
#include <hip/hip_runtime.h>
#include <hip/hip_bf16.h>

typedef __attribute__((ext_vector_type(16))) _Float16 v16h;
typedef __attribute__((ext_vector_type(8)))  float    v8f;

#define VQ_C      64
#define VQ_HW     4096
#define VQ_N      131072      /* 32 * 64 * 64 positions */
#define VQ_K      1024
#define VQ_QELEMS 8388608     /* 32*64*64*64 */

// ---------------- kernel 0: codebook prep ----------------
// c2[k] = ||codebook[k]||^2 (fp32), cbh = f16 copy of codebook, zero histogram.
__global__ __launch_bounds__(256) void vq_prep(const float* __restrict__ cb,
                                               float* __restrict__ c2,
                                               _Float16* __restrict__ cbh,
                                               int* __restrict__ counts) {
  int k = blockIdx.x * 256 + threadIdx.x;
  if (k >= VQ_K) return;
  const float* row = cb + (size_t)k * VQ_C;
  _Float16* rowh = cbh + (size_t)k * VQ_C;
  float s = 0.f;
  #pragma unroll
  for (int d = 0; d < VQ_C; ++d) {
    float v = row[d];
    s += v * v;
    rowh[d] = (_Float16)v;
  }
  c2[k] = s;
  counts[k] = 0;
}

// ---------------- kernel 1: WMMA distance argmax ----------------
// One wave handles TWO 16-position row tiles (32 positions): each B column
// tile load feeds 4 WMMAs (two independent accumulator chains), halving
// codebook traffic and keeping the XDL pipe busy. B is double-buffered so
// next-tile loads overlap current-tile WMMAs + argmax compares.
// score = 2*x.c - ||c||^2 via v_wmma_f32_16x16x32_f16, C init = -||c||^2.
__global__ __launch_bounds__(256) void vq_argmax(const float* __restrict__ x,
                                                 const _Float16* __restrict__ cbh,
                                                 const float* __restrict__ c2,
                                                 int* __restrict__ idxout) {
  const int lane  = threadIdx.x & 31;
  const int wave  = threadIdx.x >> 5;
  const int tile2 = (blockIdx.x * 8 + wave) << 1; // 512 blocks * 8 waves * 2 tiles
  const int m0    = tile2 << 4;                   // base position (32 rows, same batch)
  const int bb    = m0 >> 12;
  const int hw0   = m0 & 4095;
  const int M     = lane & 15;
  const int half  = lane >> 4;

  // ---- load two A tiles (16x64 each) as f16, scaled by 2.
  // 16-bit A layout: element e -> K = 16*half + e (matches contiguous B)
  const float* xp0 = x + (size_t)bb * (VQ_C * VQ_HW) + hw0 + M;
  const float* xp1 = xp0 + 16;
  v16h a0, a1, a2, a3;
  #pragma unroll
  for (int e = 0; e < 16; ++e) {
    const int i = e >> 1, j = e & 1;
    const int dk = 16 * half + e;
    a0[e] = (_Float16)(2.0f * xp0[(size_t)dk * VQ_HW]);
    a1[e] = (_Float16)(2.0f * xp0[(size_t)(dk + 32) * VQ_HW]);
    a2[e] = (_Float16)(2.0f * xp1[(size_t)dk * VQ_HW]);
    a3[e] = (_Float16)(2.0f * xp1[(size_t)(dk + 32) * VQ_HW]);
  }

  float bestS0[8], bestS1[8];
  int   bestI0[8], bestI1[8];
  #pragma unroll
  for (int v = 0; v < 8; ++v) {
    bestS0[v] = -3.402823466e38f; bestI0[v] = 0;
    bestS1[v] = -3.402823466e38f; bestI1[v] = 0;
  }

  const int ncol = lane & 15;   // C/D layout: lane%16 == column N

  // B operand: 32x16 f16; lane holds column ncol, halves split K 0-15/16-31.
  const _Float16* brow = cbh + (size_t)ncol * VQ_C + 16 * half;
  v16h b0 = *(const v16h*)(brow);        // current tile, K = 0..31 slice
  v16h b1 = *(const v16h*)(brow + 32);   // current tile, K = 32..63 slice

  for (int ct = 0; ct < 64; ++ct) {
    const int code = (ct << 4) + ncol;
    const float nc2 = -c2[code];

    // prefetch next column tile's B (wraps to ct=0 on tail: harmless, cached)
    const int ctn = (ct + 1) & 63;
    const _Float16* brown = cbh + (size_t)((ctn << 4) + ncol) * VQ_C + 16 * half;
    v16h nb0 = *(const v16h*)(brown);
    v16h nb1 = *(const v16h*)(brown + 32);

    v8f acc0, acc1;
    #pragma unroll
    for (int v = 0; v < 8; ++v) { acc0[v] = nc2; acc1[v] = nc2; }

    acc0 = __builtin_amdgcn_wmma_f32_16x16x32_f16(false, a0, false, b0,
                                                  (short)0, acc0, false, false);
    acc1 = __builtin_amdgcn_wmma_f32_16x16x32_f16(false, a2, false, b0,
                                                  (short)0, acc1, false, false);
    acc0 = __builtin_amdgcn_wmma_f32_16x16x32_f16(false, a1, false, b1,
                                                  (short)0, acc0, false, false);
    acc1 = __builtin_amdgcn_wmma_f32_16x16x32_f16(false, a3, false, b1,
                                                  (short)0, acc1, false, false);
    #pragma unroll
    for (int v = 0; v < 8; ++v) {
      if (acc0[v] > bestS0[v]) { bestS0[v] = acc0[v]; bestI0[v] = code; }
      if (acc1[v] > bestS1[v]) { bestS1[v] = acc1[v]; bestI1[v] = code; }
    }
    b0 = nb0; b1 = nb1;
  }

  // reduce across the 16 lanes holding the same M rows (xor 1,2,4,8 stays in half)
  #pragma unroll
  for (int v = 0; v < 8; ++v) {
    float s0 = bestS0[v]; int i0 = bestI0[v];
    float s1 = bestS1[v]; int i1 = bestI1[v];
    #pragma unroll
    for (int m = 1; m <= 8; m <<= 1) {
      float so0 = __shfl_xor(s0, m, 32);
      int   io0 = __shfl_xor(i0, m, 32);
      float so1 = __shfl_xor(s1, m, 32);
      int   io1 = __shfl_xor(i1, m, 32);
      if (so0 > s0 || (so0 == s0 && io0 < i0)) { s0 = so0; i0 = io0; }
      if (so1 > s1 || (so1 == s1 && io1 < i1)) { s1 = so1; i1 = io1; }
    }
    bestI0[v] = i0; bestI1[v] = i1;
  }
  if (ncol == 0) {              // lane 0 -> rows 0..7, lane 16 -> rows 8..15
    #pragma unroll
    for (int v = 0; v < 8; ++v) {
      idxout[m0 + (half << 3) + v]      = bestI0[v];
      idxout[m0 + 16 + (half << 3) + v] = bestI1[v];
    }
  }
}

// ---------------- kernel 2: gather + quantized out + SSE + histogram ----------------
__global__ __launch_bounds__(256) void vq_gather(const float* __restrict__ x,
                                                 const float* __restrict__ cb,
                                                 const int* __restrict__ idx,
                                                 float* __restrict__ qout,
                                                 float* __restrict__ bsum,
                                                 int* __restrict__ counts) {
  __shared__ float wsum[8];
  const int n = blockIdx.x * 256 + threadIdx.x;   // 0..131071
  const int b = n >> 12, hw = n & 4095;
  const int i = idx[n];
  const float* crow = cb + (size_t)i * VQ_C;
  const float* xp = x + (size_t)b * (VQ_C * VQ_HW) + hw;
  float* qp = qout + (size_t)b * (VQ_C * VQ_HW) + hw;
  float s = 0.f;
  #pragma unroll
  for (int c = 0; c < VQ_C; ++c) {
    float q = crow[c];
    float xv = xp[(size_t)c * VQ_HW];
    float d = q - xv;
    s += d * d;
    qp[(size_t)c * VQ_HW] = q;     // coalesced across threads at each c
  }
  atomicAdd(&counts[i], 1);        // integer atomics: deterministic
  // fixed-order block reduction -> deterministic fp sum
  #pragma unroll
  for (int m = 16; m >= 1; m >>= 1) s += __shfl_xor(s, m, 32);
  if ((threadIdx.x & 31) == 0) wsum[threadIdx.x >> 5] = s;
  __syncthreads();
  if (threadIdx.x == 0) {
    float t = 0.f;
    #pragma unroll
    for (int w = 0; w < 8; ++w) t += wsum[w];
    bsum[blockIdx.x] = t;
  }
}

// ---------------- kernel 3: one-hot encodings [32,1024,4096] ----------------
__global__ __launch_bounds__(256) void vq_onehot(const int* __restrict__ idx,
                                                 float* __restrict__ oh) {
  const size_t t = (size_t)blockIdx.x * 256 + threadIdx.x;
  const size_t e = t << 2;                 // 4 elements per thread
  const int b = (int)(e >> 22);            // / (1024*4096)
  const int k = (int)((e >> 12) & 1023);
  const int n = (int)(e & 4095);           // multiple of 4
  const int4 iv = *(const int4*)(idx + ((size_t)b << 12) + n);
  float4 o;
  o.x = (iv.x == k) ? 1.f : 0.f;
  o.y = (iv.y == k) ? 1.f : 0.f;
  o.z = (iv.z == k) ? 1.f : 0.f;
  o.w = (iv.w == k) ? 1.f : 0.f;
  *(float4*)(oh + e) = o;
}

// ---------------- kernel 4: loss + perplexity scalars ----------------
__global__ __launch_bounds__(1024) void vq_finalize(const float* __restrict__ bsum,
                                                    const int* __restrict__ counts,
                                                    float* __restrict__ loss_out,
                                                    float* __restrict__ perp_out) {
  __shared__ float red[32];
  const int t = threadIdx.x;               // 1024 threads == K
  float avg = (float)counts[t] * (1.0f / (float)VQ_N);
  float term = avg * logf(avg + 1e-10f);
  #pragma unroll
  for (int m = 16; m >= 1; m >>= 1) term += __shfl_xor(term, m, 32);
  if ((t & 31) == 0) red[t >> 5] = term;
  __syncthreads();
  if (t == 0) {
    float ent = 0.f;
    #pragma unroll
    for (int w = 0; w < 32; ++w) ent += red[w];
    *perp_out = expf(-ent);
    float sse = 0.f;
    for (int b = 0; b < 512; ++b) sse += bsum[b];   // fixed order: deterministic
    *loss_out = 0.25f * sse / (float)VQ_QELEMS;
  }
}

extern "C" void kernel_launch(void* const* d_in, const int* in_sizes, int n_in,
                              void* d_out, int out_size, void* d_ws, size_t ws_size,
                              hipStream_t stream) {
  const float* inputs   = (const float*)d_in[0];   // [32,64,64,64] BCHW fp32
  const float* codebook = (const float*)d_in[1];   // [1024,64] fp32

  float* out  = (float*)d_out;
  float* loss = out;                       // [0]
  float* q    = out + 1;                   // [1 .. 8388608]   BCHW quantized
  float* perp = out + 1 + VQ_QELEMS;       // [8388609]
  float* oh   = out + 2 + VQ_QELEMS;       // [8388610 ..]     [32,1024,4096]

  // workspace layout (all offsets keep >=32B alignment)
  char* ws = (char*)d_ws;
  float*    c2     = (float*)(ws);                     //  1024 f32
  int*      counts = (int*)  (ws + 4096);              //  1024 i32
  float*    bsum   = (float*)(ws + 8192);              //   512 f32 partial SSE
  _Float16* cbh    = (_Float16*)(ws + 10240);          // 65536 f16 (128 KB)
  int*      idxbuf = (int*)  (ws + 10240 + 131072);    // 131072 i32 (512 KB)

  vq_prep   <<<4,      256, 0, stream>>>(codebook, c2, cbh, counts);
  vq_argmax <<<512,    256, 0, stream>>>(inputs, cbh, c2, idxbuf);
  vq_gather <<<512,    256, 0, stream>>>(inputs, codebook, idxbuf, q, bsum, counts);
  vq_onehot <<<131072, 256, 0, stream>>>(idxbuf, oh);
  vq_finalize<<<1,    1024, 0, stream>>>(bsum, counts, loss, perp);
}
